// LBLBiLmVer1_90512140796596
// MI455X (gfx1250) — matrix operand order repaired and checked by reading, not compiled, not run on hardware
//
#include <hip/hip_runtime.h>
#include <math.h>
#include <stdint.h>

// ---------------- types ----------------
typedef __bf16 bf16_t;
typedef __attribute__((ext_vector_type(4)))  __bf16 v4bf;
typedef __attribute__((ext_vector_type(8)))  __bf16 v8bf;
typedef __attribute__((ext_vector_type(16))) __bf16 v16bf;
typedef __attribute__((ext_vector_type(4)))  float  v4f;
typedef __attribute__((ext_vector_type(8)))  float  v8f;

#define DIMD 512
#define N2   1024
#define MTOT 32768      // B*S = 32*1024
#define BM   256        // rows per block
#define LDA  72         // padded LDS row stride (elements)
#define LDB  72

// ---------------- fp32 -> bf16 weight conversion ----------------
__global__ __launch_bounds__(256) void cvt_bf16_kernel(const float* __restrict__ in,
                                                       bf16_t* __restrict__ out, int n4) {
    int i = blockIdx.x * blockDim.x + threadIdx.x;
    if (i < n4) {
        v4f v = *(const v4f*)(in + (size_t)i * 4);
        v4bf o;
        o[0] = (bf16_t)v[0]; o[1] = (bf16_t)v[1];
        o[2] = (bf16_t)v[2]; o[3] = (bf16_t)v[3];
        *(v4bf*)(out + (size_t)i * 4) = o;
    }
}

// ---------------- windowed context (left/right), fp32 in -> bf16 out ----------------
__global__ __launch_bounds__(256) void ctx_kernel(const float* __restrict__ inp,   // [32,1024,512]
                                                  const float* __restrict__ pad,   // [4,512]
                                                  const float* __restrict__ wts,   // [4]
                                                  bf16_t* __restrict__ out,        // [32768,512]
                                                  int isRight) {
    int id = blockIdx.x * blockDim.x + threadIdx.x;  // over 32*1024*128
    int d4 = (id & 127) << 2;
    int s  = (id >> 7) & 1023;
    int b  = id >> 17;
    float wv0 = wts[0], wv1 = wts[1], wv2 = wts[2], wv3 = wts[3];
    float wv[4] = {wv0, wv1, wv2, wv3};
    v4f acc = {0.f, 0.f, 0.f, 0.f};
#pragma unroll
    for (int w = 0; w < 4; ++w) {
        const float* src;
        if (isRight) {
            int p = s + 1 + w;
            src = (p < 1024) ? (inp + ((size_t)b * 1024 + p) * DIMD + d4)
                             : (pad + (size_t)(p - 1024) * DIMD + d4);
        } else {
            int p = s - 4 + w;
            src = (p >= 0) ? (inp + ((size_t)b * 1024 + p) * DIMD + d4)
                           : (pad + (size_t)(p + 4) * DIMD + d4);
        }
        v4f v = *(const v4f*)src;
        acc += wv[w] * v;
    }
    v4bf o;
    o[0] = (bf16_t)acc[0]; o[1] = (bf16_t)acc[1];
    o[2] = (bf16_t)acc[2]; o[3] = (bf16_t)acc[3];
    *(v4bf*)(out + ((size_t)b * 1024 + s) * DIMD + d4) = o;
}

// LDS byte-offset of a generic pointer to __shared__ memory: flat->LDS mapping
// truncates to addr[31:0] (CDNA5 aperture rules), so the low 32 bits are the offset.
__device__ __forceinline__ uint32_t lds_off(const void* p) {
    return (uint32_t)(uintptr_t)p;
}

__device__ __forceinline__ void async_load_b128(uint32_t ldsaddr, uint32_t goff, uint64_t base) {
    asm volatile("global_load_async_to_lds_b128 %0, %1, %2"
                 :: "v"(ldsaddr), "v"(goff), "s"(base) : "memory");
}

__device__ __forceinline__ void wait_async0() {
    asm volatile("s_wait_asynccnt 0" ::: "memory");
}

// ---------------- fused highway GEMM layer ----------------
// proj = X @ W + b ; out = sigmoid(g)*x + (1-sigmoid(g))*relu(nl)
// Block: 256 threads = 8 waves. Tile: BM=256 rows x 32 out-cols
// (proj cols {j..j+32} U {512+j..512+j+32} -> same-lane nl/gate pairs).
// Double-buffered K pipeline using GLOBAL_LOAD_ASYNC_TO_LDS_B128 (ASYNCcnt).
__global__ __launch_bounds__(256) void highway_gemm(const bf16_t* __restrict__ X,   // [MTOT,512]
                                                    const bf16_t* __restrict__ Wb,  // [512,1024]
                                                    const float* __restrict__ bias, // [1024]
                                                    bf16_t* __restrict__ Ybf,       // [MTOT,512] (if !isFinal)
                                                    float* __restrict__ out0,
                                                    float* __restrict__ out1,
                                                    int colOff, int isFinal) {
    __shared__ bf16_t Alds[2][BM * LDA];
    __shared__ bf16_t Blds[2][64 * LDB];

    const int tid  = threadIdx.x;
    const int wave = tid >> 5;         // 0..7
    const int lane = tid & 31;
    const int half = lane >> 4;
    const int ln16 = lane & 15;
    const int mBlk = blockIdx.x * BM;  // 128 blocks
    const int j    = blockIdx.y * 32;  // 16 blocks

    const uint64_t xg = (uint64_t)(uintptr_t)X;
    const uint64_t wg = (uint64_t)(uintptr_t)Wb;

    // issue one K-chunk (64 wide) of async global->LDS copies into buffer `buf`
    auto issue = [&](int buf, int kc) {
        // A chunk: 256 rows x 64 cols bf16 (2048 x 16B segments, 8 per thread)
#pragma unroll
        for (int i = 0; i < 8; ++i) {
            int s = tid + i * 256;
            int row = s >> 3, seg = s & 7;
            uint32_t la = lds_off(&Alds[buf][row * LDA + seg * 8]);
            uint32_t go = (uint32_t)(((mBlk + row) * DIMD + kc + seg * 8) * 2);
            async_load_b128(la, go, xg);
        }
        // B chunk: 64 k-rows x 64 n-cols (nl half then gate half), 2 per thread
#pragma unroll
        for (int i = 0; i < 2; ++i) {
            int s = tid + i * 256;
            int row = s >> 3, seg = s & 7;
            int n = (seg < 4) ? (j + seg * 8) : (512 + j + (seg - 4) * 8);
            uint32_t la = lds_off(&Blds[buf][row * LDB + seg * 8]);
            uint32_t go = (uint32_t)(((kc + row) * N2 + n) * 2);
            async_load_b128(la, go, wg);
        }
    };

    v8f acc[2][4];
    const v8f vz = {0.f, 0.f, 0.f, 0.f, 0.f, 0.f, 0.f, 0.f};
#pragma unroll
    for (int u = 0; u < 2; ++u)
#pragma unroll
        for (int t = 0; t < 4; ++t) acc[u][t] = vz;

    issue(0, 0);

    for (int kc = 0; kc < DIMD; kc += 64) {
        const int cur = (kc >> 6) & 1;
        wait_async0();        // this wave's async LDS writes for buf[cur] have landed
        __syncthreads();      // => everyone's writes landed; prior reads of buf[cur^1] done
        if (kc + 64 < DIMD) issue(cur ^ 1, kc + 64);

        const bf16_t* Ab = Alds[cur];
        const bf16_t* Bb = Blds[cur];
#pragma unroll
        for (int ks = 0; ks < 64; ks += 32) {
            // A fragments: lane = M row, two contiguous 16B chunks per lane
            v16bf a[2];
#pragma unroll
            for (int u = 0; u < 2; ++u) {
                const bf16_t* ap = Ab + (wave * 32 + u * 16 + ln16) * LDA + ks + half * 8;
                v8bf alo = *(const v8bf*)ap;
                v8bf ahi = *(const v8bf*)(ap + 16);
                a[u] = __builtin_shufflevector(alo, ahi, 0, 1, 2, 3, 4, 5, 6, 7,
                                               8, 9, 10, 11, 12, 13, 14, 15);
            }
            // B fragments: lane = N column, 16 K values per lane (table order)
            const bf16_t* bp = Bb + (ks + half * 8) * LDB + ln16;
#pragma unroll
            for (int t = 0; t < 4; ++t) {
                v16bf bb;
#pragma unroll
                for (int e = 0; e < 16; ++e)
                    bb[e] = bp[(e < 8 ? e : e + 8) * LDB + t * 16];
#pragma unroll
                for (int u = 0; u < 2; ++u) {
                    acc[u][t] = __builtin_amdgcn_wmma_f32_16x16x32_bf16(
                        false, a[u], false, bb, (short)0, acc[u][t], false, false);
                }
            }
        }
    }

    // fused epilogue: bias + sigmoid gate + highway combine, all in registers
#pragma unroll
    for (int u = 0; u < 2; ++u) {
#pragma unroll
        for (int t = 0; t < 2; ++t) {
            int col = j + t * 16 + ln16;            // output D-column
            float bn = bias[col];
            float bg = bias[512 + col];
#pragma unroll
            for (int v = 0; v < 8; ++v) {
                int m = mBlk + wave * 32 + u * 16 + v + 8 * half;
                float nl = acc[u][t][v] + bn;
                float g  = acc[u][t + 2][v] + bg;
                float sg = 1.0f / (1.0f + __expf(-g));
                float xv = (float)X[(size_t)m * DIMD + col];
                float r  = sg * xv + (1.0f - sg) * fmaxf(nl, 0.0f);
                if (isFinal) {
                    size_t o = (size_t)m * N2 + colOff + col;
                    out0[o] = r;
                    out1[o] = r;
                } else {
                    Ybf[(size_t)m * DIMD + col] = (bf16_t)r;
                }
            }
        }
    }
}

// ---------------- launcher ----------------
extern "C" void kernel_launch(void* const* d_in, const int* in_sizes, int n_in,
                              void* d_out, int out_size, void* d_ws, size_t ws_size,
                              hipStream_t stream) {
    const float* inputs = (const float*)d_in[0];
    const float* lpad   = (const float*)d_in[1];
    const float* rpad   = (const float*)d_in[2];
    const float* lwts   = (const float*)d_in[3];
    const float* rwts   = (const float*)d_in[4];
    const float* lW     = (const float*)d_in[5];
    const float* lb     = (const float*)d_in[6];
    const float* rW     = (const float*)d_in[7];
    const float* rb     = (const float*)d_in[8];

    float* out0 = (float*)d_out;
    float* out1 = out0 + (size_t)MTOT * N2;  // second tuple element (identical data)

    char* ws = (char*)d_ws;
    bf16_t* bufA = (bf16_t*)ws;                                  // 32 MB
    bf16_t* bufB = (bf16_t*)(ws + (size_t)MTOT * DIMD * 2);      // 32 MB
    bf16_t* WbfL = (bf16_t*)(ws + (size_t)MTOT * DIMD * 4);      // 2 MB
    bf16_t* WbfR = WbfL + (size_t)2 * 512 * 1024;                // 2 MB

    const int nW4 = (2 * 512 * 1024) / 4;  // 262144
    cvt_bf16_kernel<<<(nW4 + 255) / 256, 256, 0, stream>>>(lW, WbfL, nW4);
    cvt_bf16_kernel<<<(nW4 + 255) / 256, 256, 0, stream>>>(rW, WbfR, nW4);

    dim3 gCtx(16384), bCtx(256);
    dim3 gG(MTOT / BM, 16), bG(256);

    // left stack
    ctx_kernel<<<gCtx, bCtx, 0, stream>>>(inputs, lpad, lwts, bufA, 0);
    highway_gemm<<<gG, bG, 0, stream>>>(bufA, WbfL, lb, bufB, nullptr, nullptr, 0, 0);
    highway_gemm<<<gG, bG, 0, stream>>>(bufB, WbfL + 512 * 1024, lb + 1024,
                                        nullptr, out0, out1, 0, 1);
    // right stack
    ctx_kernel<<<gCtx, bCtx, 0, stream>>>(inputs, rpad, rwts, bufA, 1);
    highway_gemm<<<gG, bG, 0, stream>>>(bufA, WbfR, rb, bufB, nullptr, nullptr, 0, 0);
    highway_gemm<<<gG, bG, 0, stream>>>(bufB, WbfR + 512 * 1024, rb + 1024,
                                        nullptr, out0, out1, 512, 1);
}
